// WindowedAttentionCore_59674275611314
// MI455X (gfx1250) — matrix operand order repaired and checked by reading, not compiled, not run on hardware
//
#include <hip/hip_runtime.h>

// ---------------------------------------------------------------------------
// MI455X (gfx1250) windowed GQA attention, bf16 WMMA everywhere.
// Roofline: ~55 GFLOP vs ~80MB HBM traffic -> matrix-unit bound; all
// intermediates live in the 192MB L2. wave32, 16x16x32 bf16 WMMA, f32 accum.
// GEMMs use double-buffered GLOBAL_LOAD_ASYNC_TO_LDS_B128 staging (ASYNCcnt).
// ---------------------------------------------------------------------------

typedef unsigned short u16;
typedef __attribute__((ext_vector_type(16))) __bf16 v16bf;
typedef __attribute__((ext_vector_type(8)))  float  v8f;

#define SEQ    2048
#define HIDDEN 2048
#define NHEADS 16
#define HD     128
#define KVDIM  512   // NUM_KV_HEADS * HEAD_DIM

union Frag {      // 16 bf16 = 32 bytes = 2x uint4
  uint4 u[2];
  v16bf v;
};

__device__ inline u16 f32_to_bf16(float f) {
  unsigned int u = __float_as_uint(f);
  u += 0x7fffu + ((u >> 16) & 1u);   // round-to-nearest-even
  return (u16)(u >> 16);
}

__device__ inline v8f zero8() {
  v8f z;
#pragma unroll
  for (int i = 0; i < 8; i++) z[i] = 0.0f;
  return z;
}

// Async copy 32B (two B128 ops, ISA adds INST_OFFSET to both LDS and global
// addresses) directly into LDS; tracked by ASYNCcnt, no VGPR data movement.
__device__ inline void async_cp32B(const u16* g, const u16* l) {
  unsigned lo = (unsigned)(size_t)l;                 // low 32 bits = LDS offset
  unsigned long long ga = (unsigned long long)(size_t)g;
  asm volatile("global_load_async_to_lds_b128 %0, %1, off"
               :: "v"(lo), "v"(ga) : "memory");
  asm volatile("global_load_async_to_lds_b128 %0, %1, off offset:16"
               :: "v"(lo), "v"(ga) : "memory");
}

// ---------------- elementwise f32 -> bf16 ----------------
__global__ __launch_bounds__(256) void cvt_bf16_kernel(const float* __restrict__ in,
                                                       u16* __restrict__ out, int n) {
  int id = blockIdx.x * 256 + threadIdx.x;
  if (id < n) out[id] = f32_to_bf16(in[id]);
}

// ---------------- weight convert + transpose: W[k][n] -> WT[n][k] ----------
__global__ __launch_bounds__(256) void cvt_wT_kernel(const float* __restrict__ W,
                                                     u16* __restrict__ WT, int K, int N) {
  int id = blockIdx.x * 256 + threadIdx.x;   // id = n*K + k  (write-coalesced)
  if (id >= K * N) return;
  int k = id % K;
  int n = id / K;
  WT[id] = f32_to_bf16(W[(size_t)k * N + n]);
}

// ---------------- LDS-tiled bf16 GEMM: C[M,N] = A[M,K] * BT[N,K]^T ---------
// 128x128 block tile, 8 waves x (32x64), K-step 32, double-buffered async LDS.
// MODE 0: f32 out.  MODE 1: bf16 out.  MODE 2: bf16 out, transposed (C[n][m]).
template <int MODE>
__global__ __launch_bounds__(256) void gemm_bf16_kernel(
    const u16* __restrict__ A, const u16* __restrict__ BT,
    void* __restrict__ Cp, int M, int N, int K) {
  __shared__ u16 lA[2][128][40];   // 32 halves + 8 pad, keeps 16B alignment
  __shared__ u16 lB[2][128][40];

  const int tid  = threadIdx.x;
  const int lane = tid & 31;
  const int wave = tid >> 5;
  const int lr   = lane & 15;
  const int g    = lane >> 4;
  const int m0   = blockIdx.y * 128;
  const int n0   = blockIdx.x * 128;
  const int wm   = wave >> 1;   // 0..3  -> 32 rows each
  const int wn   = wave & 1;    // 0..1  -> 64 cols each

  v8f acc[2][4];
#pragma unroll
  for (int i = 0; i < 2; i++)
#pragma unroll
    for (int j = 0; j < 4; j++) acc[i][j] = zero8();

  const int ldr = tid >> 1;         // 0..127 staging row
  const int ldc = (tid & 1) << 4;   // 0 or 16

  // prologue: stage k0=0 into buffer 0 (4 async B128 per thread per stage)
  async_cp32B(A  + (size_t)(m0 + ldr) * K + ldc, &lA[0][ldr][ldc]);
  async_cp32B(BT + (size_t)(n0 + ldr) * K + ldc, &lB[0][ldr][ldc]);

  for (int k0 = 0; k0 < K; k0 += 32) {
    const int cur = (k0 >> 5) & 1;
    if (k0 + 32 < K) {  // prefetch next K-step into the other buffer
      async_cp32B(A  + (size_t)(m0 + ldr) * K + (k0 + 32) + ldc, &lA[cur ^ 1][ldr][ldc]);
      async_cp32B(BT + (size_t)(n0 + ldr) * K + (k0 + 32) + ldc, &lB[cur ^ 1][ldr][ldc]);
      // async ops complete in order per wave: <=4 outstanding means the
      // current buffer's 4 ops have landed, next stage still in flight.
      asm volatile("s_wait_asynccnt 0x4" ::: "memory");
    } else {
      asm volatile("s_wait_asynccnt 0x0" ::: "memory");
    }
    __syncthreads();

    Frag fa[2], fb[4];
#pragma unroll
    for (int i = 0; i < 2; i++) {    // A frag: lane=row, K groups of 8 per half
      int r = wm * 32 + i * 16 + lr;
      fa[i].u[0] = *(const uint4*)&lA[cur][r][g * 8];
      fa[i].u[1] = *(const uint4*)&lA[cur][r][16 + g * 8];
    }
#pragma unroll
    for (int j = 0; j < 4; j++) {    // B frag: lane=col, K groups of 8 per half
      int r = wn * 64 + j * 16 + lr;
      fb[j].u[0] = *(const uint4*)&lB[cur][r][g * 8];
      fb[j].u[1] = *(const uint4*)&lB[cur][r][16 + g * 8];
    }
#pragma unroll
    for (int i = 0; i < 2; i++)
#pragma unroll
      for (int j = 0; j < 4; j++)
        acc[i][j] = __builtin_amdgcn_wmma_f32_16x16x32_bf16(
            false, fa[i].v, false, fb[j].v, (short)0, acc[i][j], false, false);
    __syncthreads();   // all waves done reading before this buffer is re-staged
  }

#pragma unroll
  for (int i = 0; i < 2; i++)
#pragma unroll
    for (int j = 0; j < 4; j++)
#pragma unroll
      for (int r = 0; r < 8; r++) {
        int row = m0 + wm * 32 + i * 16 + r + 8 * g;
        int col = n0 + wn * 64 + j * 16 + lr;
        float v = acc[i][j][r];
        if (MODE == 0)
          ((float*)Cp)[(size_t)row * N + col] = v;
        else if (MODE == 1)
          ((u16*)Cp)[(size_t)row * N + col] = f32_to_bf16(v);
        else
          ((u16*)Cp)[(size_t)col * M + row] = f32_to_bf16(v);
      }
}

// ---------------- windowed GQA flash-attention core ------------------------
// One wave handles one (head, 16-query tile). Adaptive window <= 512 keys, so
// we only walk key tiles in [i0-510, i0+15]; the bool mask supplies exactness.
__global__ __launch_bounds__(256) void attn_kernel(
    const u16* __restrict__ Qb,            // [S, HIDDEN] bf16
    const u16* __restrict__ Kb,            // [S, KVDIM]  bf16
    const u16* __restrict__ VT,            // [KVDIM, S]  bf16 (transposed V)
    const unsigned char* __restrict__ msk, // [S, S] bool
    u16* __restrict__ Ob)                  // [S, HIDDEN] bf16
{
  __shared__ u16 lP[8][16][40];  // per-wave P restaging (C-layout -> A-layout)
  const int lane = threadIdx.x & 31;
  const int wave = threadIdx.x >> 5;
  const int lr   = lane & 15;
  const int g    = lane >> 4;
  const int h    = blockIdx.y;
  const int i0   = (blockIdx.x * 8 + wave) * 16;
  const int kvh  = h >> 2;  // GQA: 4 query heads per KV head
  const float scale = 0.08838834764831845f;  // 1/sqrt(128)

  // Q fragments for the whole head_dim (4 K-chunks of 32)
  Frag fq[4];
  const u16* qp = Qb + (size_t)(i0 + lr) * HIDDEN + h * HD;
#pragma unroll
  for (int kc = 0; kc < 4; kc++) {
    fq[kc].u[0] = *(const uint4*)(qp + kc * 32 + g * 8);
    fq[kc].u[1] = *(const uint4*)(qp + kc * 32 + 16 + g * 8);
  }

  v8f oacc[8];  // 16 rows x 128 d
#pragma unroll
  for (int t = 0; t < 8; t++) oacc[t] = zero8();
  float rowM[8], rowL[8];
#pragma unroll
  for (int r = 0; r < 8; r++) { rowM[r] = -3.0e38f; rowL[r] = 0.0f; }

  int jstart = i0 - 510;           // win < 512 -> key >= i-510
  if (jstart < 0) jstart = 0;
  jstart &= ~31;

  for (int j0 = jstart; j0 <= i0 + 15; j0 += 32) {
    // ---- scores: Q(16x128) . K_tile^T -> two 16x16 f32 tiles ----
    v8f sc[2];
    sc[0] = zero8();
    sc[1] = zero8();
#pragma unroll
    for (int sub = 0; sub < 2; sub++) {
      const u16* kp = Kb + (size_t)(j0 + sub * 16 + lr) * KVDIM + kvh * HD + g * 8;
#pragma unroll
      for (int kc = 0; kc < 4; kc++) {
        Frag fk;
        fk.u[0] = *(const uint4*)(kp + kc * 32);
        fk.u[1] = *(const uint4*)(kp + kc * 32 + 16);
        sc[sub] = __builtin_amdgcn_wmma_f32_16x16x32_bf16(
            false, fq[kc].v, false, fk.v, (short)0, sc[sub], false, false);
      }
    }
    // ---- scale + mask ----
#pragma unroll
    for (int sub = 0; sub < 2; sub++) {
      int jc = j0 + sub * 16 + lr;
      const unsigned char* mrow = msk + (size_t)(i0 + 8 * g) * SEQ + jc;
#pragma unroll
      for (int r = 0; r < 8; r++) {
        float s = sc[sub][r] * scale;
        sc[sub][r] = mrow[(size_t)r * SEQ] ? s : -3.0e38f;
      }
    }
    // ---- online softmax (row stats across 16-lane halves) ----
#pragma unroll
    for (int r = 0; r < 8; r++) {
      float mx = fmaxf(sc[0][r], sc[1][r]);
#pragma unroll
      for (int off = 8; off > 0; off >>= 1) mx = fmaxf(mx, __shfl_xor(mx, off, 16));
      float mnew = fmaxf(rowM[r], mx);
      float corr = __expf(rowM[r] - mnew);
      rowM[r] = mnew;
      float p0 = __expf(sc[0][r] - mnew);
      float p1 = __expf(sc[1][r] - mnew);
      float ps = p0 + p1;
#pragma unroll
      for (int off = 8; off > 0; off >>= 1) ps += __shfl_xor(ps, off, 16);
      rowL[r] = rowL[r] * corr + ps;
#pragma unroll
      for (int t = 0; t < 8; t++) oacc[t][r] *= corr;
      lP[wave][r + 8 * g][lr]      = f32_to_bf16(p0);   // C-layout -> LDS
      lP[wave][r + 8 * g][16 + lr] = f32_to_bf16(p1);
    }
    // wave-private restage: only need DS-counter drain, no barrier (CDNA5
    // split counters: s_wait_dscnt)
    asm volatile("s_wait_dscnt 0" ::: "memory");
    Frag fp;  // P as 16x32 A-fragment
    fp.u[0] = *(const uint4*)&lP[wave][lr][g * 8];
    fp.u[1] = *(const uint4*)&lP[wave][lr][16 + g * 8];
    // ---- oacc += P(16x32) . V_tile(32x128) ----
#pragma unroll
    for (int t = 0; t < 8; t++) {
      const u16* vp = VT + (size_t)(kvh * HD + t * 16 + lr) * SEQ + j0 + g * 8;
      Frag fv;
      fv.u[0] = *(const uint4*)vp;
      fv.u[1] = *(const uint4*)(vp + 16);
      oacc[t] = __builtin_amdgcn_wmma_f32_16x16x32_bf16(
          false, fp.v, false, fv.v, (short)0, oacc[t], false, false);
    }
  }

  // ---- normalize + store bf16 for the Wo projection ----
  u16* op = Ob + (size_t)(i0 + 8 * g) * HIDDEN + h * HD + lr;
#pragma unroll
  for (int t = 0; t < 8; t++)
#pragma unroll
    for (int r = 0; r < 8; r++)
      op[(size_t)r * HIDDEN + t * 16] = f32_to_bf16(oacc[t][r] / rowL[r]);
}

// ---------------------------------------------------------------------------
extern "C" void kernel_launch(void* const* d_in, const int* in_sizes, int n_in,
                              void* d_out, int out_size, void* d_ws, size_t ws_size,
                              hipStream_t stream) {
  (void)in_sizes; (void)n_in; (void)out_size; (void)ws_size;
  const float* X            = (const float*)d_in[0];
  const unsigned char* mask = (const unsigned char*)d_in[1];  // bool mask
  const float* Wq           = (const float*)d_in[2];
  const float* Wk           = (const float*)d_in[3];
  const float* Wv           = (const float*)d_in[4];
  const float* Wo           = (const float*)d_in[5];
  float* out                = (float*)d_out;

  char* ws = (char*)d_ws;
  size_t off = 0;
  u16* Xb  = (u16*)(ws + off); off += (size_t)SEQ * HIDDEN * 2;     // X bf16
  u16* WqT = (u16*)(ws + off); off += (size_t)HIDDEN * HIDDEN * 2;  // Wq^T bf16
  u16* WkT = (u16*)(ws + off); off += (size_t)KVDIM * HIDDEN * 2;   // Wk^T bf16
  u16* WvT = (u16*)(ws + off); off += (size_t)KVDIM * HIDDEN * 2;   // Wv^T bf16
  u16* WoT = (u16*)(ws + off); off += (size_t)HIDDEN * HIDDEN * 2;  // Wo^T bf16
  u16* Qb  = (u16*)(ws + off); off += (size_t)SEQ * HIDDEN * 2;     // Q bf16
  u16* Kb  = (u16*)(ws + off); off += (size_t)SEQ * KVDIM * 2;      // K bf16
  u16* Vt  = (u16*)(ws + off); off += (size_t)KVDIM * SEQ * 2;      // V^T bf16
  u16* Ob  = (u16*)(ws + off); off += (size_t)SEQ * HIDDEN * 2;     // attn out bf16

  dim3 blk(256);
  cvt_bf16_kernel<<<SEQ * HIDDEN / 256, blk, 0, stream>>>(X, Xb, SEQ * HIDDEN);
  cvt_wT_kernel<<<HIDDEN * HIDDEN / 256, blk, 0, stream>>>(Wq, WqT, HIDDEN, HIDDEN);
  cvt_wT_kernel<<<HIDDEN * KVDIM / 256, blk, 0, stream>>>(Wk, WkT, HIDDEN, KVDIM);
  cvt_wT_kernel<<<HIDDEN * KVDIM / 256, blk, 0, stream>>>(Wv, WvT, HIDDEN, KVDIM);
  cvt_wT_kernel<<<HIDDEN * HIDDEN / 256, blk, 0, stream>>>(Wo, WoT, HIDDEN, HIDDEN);

  // Projections
  gemm_bf16_kernel<1><<<dim3(HIDDEN / 128, SEQ / 128), blk, 0, stream>>>(
      Xb, WqT, Qb, SEQ, HIDDEN, HIDDEN);
  gemm_bf16_kernel<1><<<dim3(KVDIM / 128, SEQ / 128), blk, 0, stream>>>(
      Xb, WkT, Kb, SEQ, KVDIM, HIDDEN);
  gemm_bf16_kernel<2><<<dim3(KVDIM / 128, SEQ / 128), blk, 0, stream>>>(
      Xb, WvT, Vt, SEQ, KVDIM, HIDDEN);  // V stored transposed for P.V

  // Windowed attention core: 128 q-tiles x 16 heads, 8 waves per block
  attn_kernel<<<dim3(SEQ / 16 / 8, NHEADS), blk, 0, stream>>>(Qb, Kb, Vt, mask, Ob);

  // Output projection (f32 result)
  gemm_bf16_kernel<0><<<dim3(HIDDEN / 128, SEQ / 128), blk, 0, stream>>>(
      Ob, WoT, out, SEQ, HIDDEN, HIDDEN);
}
